// MotionCompensator_13829794693635
// MI455X (gfx1250) — matrix operand order repaired and checked by reading, not compiled, not run on hardware
//
#include <hip/hip_runtime.h>
#include <math.h>

typedef __attribute__((ext_vector_type(16))) __bf16 v16bf;
typedef __attribute__((ext_vector_type(8)))  float  v8f;

union Frag16 { uint4 q[2]; v16bf v; };

#define BB   2
#define CC   64
#define HH   128
#define WW   128
#define HP   130
#define WP   130
#define HWD  (HH*WW)

static __device__ __forceinline__ unsigned short f32_to_bf16(float f) {
    unsigned int u = __float_as_uint(f);
    unsigned int lsb = (u >> 16) & 1u;
    u += 0x7fffu + lsb;                 // round-to-nearest-even
    return (unsigned short)(u >> 16);
}

// ---------------------------------------------------------------- zero fill
__global__ __launch_bounds__(256) void zero32_kernel(unsigned int* p, long n) {
    long i = (long)blockIdx.x * blockDim.x + threadIdx.x;
    long stride = (long)gridDim.x * blockDim.x;
    for (; i < n; i += stride) p[i] = 0u;
}

// ---------------------------------------------------- prev -> NHWC f32 + X1 hi
__global__ __launch_bounds__(256) void pack_prev_kernel(const float* __restrict__ prev,
                                                        float* __restrict__ Xp,
                                                        unsigned short* __restrict__ X1) {
    int tid = blockIdx.x * 256 + threadIdx.x;         // B*H*W = 32768
    int x = tid & 127, y = (tid >> 7) & 127, bi = tid >> 14;
    size_t pix = ((size_t)bi * HH + y) * WW + x;
    float* xp = Xp + pix * CC;
    unsigned short* x1 = X1 + (((size_t)bi * HP + y + 1) * WP + (x + 1)) * 128 + 64;
    #pragma unroll 4
    for (int c = 0; c < CC; ++c) {
        float v = prev[((size_t)bi * CC + c) * HWD + y * WW + x];
        xp[c] = v;
        x1[c] = f32_to_bf16(v);
    }
}

// ---------------------------------------------- weight packing into B-fragments
// B-fragment layout (16-bit B, KxN=32x16): lane = n + 16*h, elem e<8: k = 8h+e, e>=8: k = 16+8h+(e-8)
__global__ __launch_bounds__(256) void pack_w1_kernel(const float* __restrict__ w1,
                                                      unsigned short* __restrict__ W1F) {
    int tid = blockIdx.x * 256 + threadIdx.x;         // 9*4*4*32 = 4608
    if (tid >= 9 * 4 * 4 * 32) return;
    int lane = tid & 31, nb = (tid >> 5) & 3, cb = (tid >> 7) & 3, t9 = tid >> 9;
    int half = lane >> 4, n = lane & 15;
    int cout = nb * 16 + n;
    int ky = t9 / 3, kx = t9 % 3;
    unsigned short* dst = W1F + ((((size_t)t9 * 4 + cb) * 4 + nb) * 32 + lane) * 16;
    #pragma unroll
    for (int e = 0; e < 16; ++e) {
        int cin = cb * 32 + (e < 8 ? half * 8 + e : 16 + half * 8 + (e - 8));
        float v = w1[(((size_t)cout * 128 + cin) * 3 + ky) * 3 + kx];
        dst[e] = f32_to_bf16(v);
    }
}

__global__ __launch_bounds__(256) void pack_w2_kernel(const float* __restrict__ w2,
                                                      unsigned short* __restrict__ W2F) {
    int tid = blockIdx.x * 256 + threadIdx.x;         // 9*2*4*32 = 2304
    if (tid >= 9 * 2 * 4 * 32) return;
    int lane = tid & 31, nb = (tid >> 5) & 3, cb = (tid >> 7) & 1, t9 = tid >> 8;
    int half = lane >> 4, n = lane & 15;
    int cout = nb * 16 + n;
    int ky = t9 / 3, kx = t9 % 3;
    unsigned short* dst = W2F + ((((size_t)t9 * 2 + cb) * 4 + nb) * 32 + lane) * 16;
    #pragma unroll
    for (int e = 0; e < 16; ++e) {
        int cin = cb * 32 + (e < 8 ? half * 8 + e : 16 + half * 8 + (e - 8));
        float v = w2[(((size_t)cout * 64 + cin) * 3 + ky) * 3 + kx];
        dst[e] = f32_to_bf16(v);
    }
}

// ------------------------------------------------- deformable conv (fp32 VALU)
__global__ __launch_bounds__(256) void dcn_kernel(const float* __restrict__ Xp,
                                                  const float* __restrict__ offs,
                                                  const float* __restrict__ dcnw,
                                                  const float* __restrict__ dcnb,
                                                  float* __restrict__ aligned,
                                                  unsigned short* __restrict__ X1) {
    int tid = blockIdx.x * 256 + threadIdx.x;         // B*8*H*W = 262144
    int x = tid & 127, y = (tid >> 7) & 127, g = (tid >> 14) & 7, bi = tid >> 17;

    float accum[8];
    #pragma unroll
    for (int o = 0; o < 8; ++o) accum[o] = 0.f;

    const float* xb = Xp + (size_t)bi * HWD * CC + g * 8;

    #pragma unroll
    for (int k = 0; k < 9; ++k) {
        int ky = k / 3, kx = k % 3;
        size_t ob = ((size_t)bi * 144 + (size_t)(g * 9 + k) * 2) * HWD + (size_t)y * WW + x;
        float dy = offs[ob];
        float dx = offs[ob + HWD];
        float py = (float)(y - 1 + ky) + dy;
        float px = (float)(x - 1 + kx) + dx;
        float y0f = floorf(py), x0f = floorf(px);
        float wy = py - y0f, wx = px - x0f;
        int y0 = (int)y0f, x0 = (int)x0f;

        float cw[4] = { (1.f - wy) * (1.f - wx), (1.f - wy) * wx,
                        wy * (1.f - wx),         wy * wx };
        int ys[4] = { y0, y0, y0 + 1, y0 + 1 };
        int xs[4] = { x0, x0 + 1, x0, x0 + 1 };

        float s[8];
        #pragma unroll
        for (int i = 0; i < 8; ++i) s[i] = 0.f;

        #pragma unroll
        for (int c4 = 0; c4 < 4; ++c4) {
            int yi = ys[c4], xi = xs[c4];
            bool valid = (yi >= 0) & (yi < HH) & (xi >= 0) & (xi < WW);
            int yc = yi < 0 ? 0 : (yi > HH - 1 ? HH - 1 : yi);
            int xc = xi < 0 ? 0 : (xi > WW - 1 ? WW - 1 : xi);
            float wgt = valid ? cw[c4] : 0.f;
            const float4* p = reinterpret_cast<const float4*>(xb + ((size_t)yc * WW + xc) * CC);
            float4 a0 = p[0], a1 = p[1];
            s[0] += wgt * a0.x; s[1] += wgt * a0.y; s[2] += wgt * a0.z; s[3] += wgt * a0.w;
            s[4] += wgt * a1.x; s[5] += wgt * a1.y; s[6] += wgt * a1.z; s[7] += wgt * a1.w;
        }
        #pragma unroll
        for (int o = 0; o < 8; ++o) {
            const float* wp = dcnw + (size_t)(g * 8 + o) * 8 * 9 + k;
            float a = 0.f;
            #pragma unroll
            for (int i = 0; i < 8; ++i) a += wp[i * 9] * s[i];
            accum[o] += a;
        }
    }

    union { unsigned short h[8]; uint4 q; } pk;
    #pragma unroll
    for (int o = 0; o < 8; ++o) {
        float v = accum[o] + dcnb[g * 8 + o];
        aligned[(((size_t)bi * CC + g * 8 + o) * HH + y) * WW + x] = v;
        pk.h[o] = f32_to_bf16(v);
    }
    *reinterpret_cast<uint4*>(X1 + (((size_t)bi * HP + y + 1) * WP + (x + 1)) * 128 + g * 8) = pk.q;
}

// ------------------------------------------------------- conv1: 128->64, WMMA
// Per wave: M=32 pixels x N=64 couts, K = 9 taps * 128 ch.
// Double-buffered fragments: prefetch iter i+1 while WMMAs of iter i execute.
__global__ __launch_bounds__(256) void conv1_wmma_kernel(const unsigned short* __restrict__ X1,
                                                         const unsigned short* __restrict__ W1F,
                                                         const float* __restrict__ b1,
                                                         unsigned short* __restrict__ X2) {
    int wave = threadIdx.x >> 5;
    int lane = threadIdx.x & 31;
    int t = blockIdx.x * 8 + wave;                    // 1024 tiles (b, y, x/32)
    int bi  = t >> 9;
    int rem = t & 511;
    int y   = rem >> 2;
    int x0  = (rem & 3) << 5;
    int half = lane >> 4, n = lane & 15;

    v8f zero = {0.f, 0.f, 0.f, 0.f, 0.f, 0.f, 0.f, 0.f};
    v8f acc[8];                                        // [mt][nb] -> acc[mt*4+nb]
    #pragma unroll
    for (int i = 0; i < 8; ++i) acc[i] = zero;

    // iter = t9*4 + cb  (36 iterations)
    auto loadIter = [&](int iter, Frag16& a0, Frag16& a1, Frag16* bf) {
        int t9 = iter >> 2, cb = iter & 3;
        int ky = t9 / 3, kx = t9 % 3;
        const unsigned short* arow =
            X1 + (((size_t)bi * HP + (y + ky)) * WP + (x0 + kx + n)) * 128;
        int c0 = cb * 32 + half * 8;                   // A layout: K {8h..8h+7, 8h+16..8h+23}
        a0.q[0] = *reinterpret_cast<const uint4*>(arow + c0);
        a0.q[1] = *reinterpret_cast<const uint4*>(arow + c0 + 16);
        a1.q[0] = *reinterpret_cast<const uint4*>(arow + 16 * 128 + c0);
        a1.q[1] = *reinterpret_cast<const uint4*>(arow + 16 * 128 + c0 + 16);
        const unsigned short* bbase =
            W1F + ((size_t)(t9 * 4 + cb) * 4) * 512 + (size_t)lane * 16;
        #pragma unroll
        for (int nb = 0; nb < 4; ++nb) {
            const uint4* bp = reinterpret_cast<const uint4*>(bbase + nb * 512);
            bf[nb].q[0] = bp[0];
            bf[nb].q[1] = bp[1];
        }
    };

    Frag16 a0c, a1c, bc[4];
    loadIter(0, a0c, a1c, bc);
    #pragma unroll
    for (int iter = 0; iter < 36; ++iter) {
        Frag16 a0n, a1n, bn[4];
        if (iter != 35) loadIter(iter + 1, a0n, a1n, bn);
        #pragma unroll
        for (int nb = 0; nb < 4; ++nb) {
            acc[nb]     = __builtin_amdgcn_wmma_f32_16x16x32_bf16(
                false, a0c.v, false, bc[nb].v, (short)0, acc[nb],     false, false);
            acc[4 + nb] = __builtin_amdgcn_wmma_f32_16x16x32_bf16(
                false, a1c.v, false, bc[nb].v, (short)0, acc[4 + nb], false, false);
        }
        if (iter != 35) {
            a0c = a0n; a1c = a1n;
            #pragma unroll
            for (int nb = 0; nb < 4; ++nb) bc[nb] = bn[nb];
        }
    }

    // epilogue: bias + lrelu -> X2 interior (padded NHWC bf16)
    #pragma unroll
    for (int mt = 0; mt < 2; ++mt) {
        #pragma unroll
        for (int nb = 0; nb < 4; ++nb) {
            int cout = nb * 16 + n;
            float bias = b1[cout];
            #pragma unroll
            for (int j = 0; j < 8; ++j) {
                int m = mt * 16 + half * 8 + j;         // C/D: m = 8*(lane>>4)+vgpr
                float v = acc[mt * 4 + nb][j] + bias;
                v = v >= 0.f ? v : 0.1f * v;
                X2[(((size_t)bi * HP + y + 1) * WP + (x0 + m + 1)) * 64 + cout] = f32_to_bf16(v);
            }
        }
    }
}

// ---------------------------------------- conv2: 64->64, WMMA + fused residual
__global__ __launch_bounds__(256) void conv2_wmma_kernel(const unsigned short* __restrict__ X2,
                                                         const unsigned short* __restrict__ W2F,
                                                         const float* __restrict__ b2,
                                                         const float* __restrict__ aligned,
                                                         float* __restrict__ out) {
    int wave = threadIdx.x >> 5;
    int lane = threadIdx.x & 31;
    int t = blockIdx.x * 8 + wave;                    // 1024 tiles
    int bi  = t >> 9;
    int rem = t & 511;
    int y   = rem >> 2;
    int x0  = (rem & 3) << 5;
    int half = lane >> 4, n = lane & 15;

    v8f zero = {0.f, 0.f, 0.f, 0.f, 0.f, 0.f, 0.f, 0.f};
    v8f acc[8];
    #pragma unroll
    for (int i = 0; i < 8; ++i) acc[i] = zero;

    // iter = t9*2 + cb  (18 iterations)
    auto loadIter = [&](int iter, Frag16& a0, Frag16& a1, Frag16* bf) {
        int t9 = iter >> 1, cb = iter & 1;
        int ky = t9 / 3, kx = t9 % 3;
        const unsigned short* arow =
            X2 + (((size_t)bi * HP + (y + ky)) * WP + (x0 + kx + n)) * 64;
        int c0 = cb * 32 + half * 8;
        a0.q[0] = *reinterpret_cast<const uint4*>(arow + c0);
        a0.q[1] = *reinterpret_cast<const uint4*>(arow + c0 + 16);
        a1.q[0] = *reinterpret_cast<const uint4*>(arow + 16 * 64 + c0);
        a1.q[1] = *reinterpret_cast<const uint4*>(arow + 16 * 64 + c0 + 16);
        const unsigned short* bbase =
            W2F + ((size_t)(t9 * 2 + cb) * 4) * 512 + (size_t)lane * 16;
        #pragma unroll
        for (int nb = 0; nb < 4; ++nb) {
            const uint4* bp = reinterpret_cast<const uint4*>(bbase + nb * 512);
            bf[nb].q[0] = bp[0];
            bf[nb].q[1] = bp[1];
        }
    };

    Frag16 a0c, a1c, bc[4];
    loadIter(0, a0c, a1c, bc);
    #pragma unroll
    for (int iter = 0; iter < 18; ++iter) {
        Frag16 a0n, a1n, bn[4];
        if (iter != 17) loadIter(iter + 1, a0n, a1n, bn);
        #pragma unroll
        for (int nb = 0; nb < 4; ++nb) {
            acc[nb]     = __builtin_amdgcn_wmma_f32_16x16x32_bf16(
                false, a0c.v, false, bc[nb].v, (short)0, acc[nb],     false, false);
            acc[4 + nb] = __builtin_amdgcn_wmma_f32_16x16x32_bf16(
                false, a1c.v, false, bc[nb].v, (short)0, acc[4 + nb], false, false);
        }
        if (iter != 17) {
            a0c = a0n; a1c = a1n;
            #pragma unroll
            for (int nb = 0; nb < 4; ++nb) bc[nb] = bn[nb];
        }
    }

    // epilogue: bias + lrelu + residual(aligned) -> NCHW f32 out
    #pragma unroll
    for (int mt = 0; mt < 2; ++mt) {
        #pragma unroll
        for (int nb = 0; nb < 4; ++nb) {
            int cout = nb * 16 + n;
            float bias = b2[cout];
            #pragma unroll
            for (int j = 0; j < 8; ++j) {
                int m = mt * 16 + half * 8 + j;
                size_t oidx = (((size_t)bi * CC + cout) * HH + y) * WW + (x0 + m);
                float v = acc[mt * 4 + nb][j] + bias;
                v = v >= 0.f ? v : 0.1f * v;
                out[oidx] = aligned[oidx] + v;
            }
        }
    }
}

// ------------------------------------------------------------------ launcher
extern "C" void kernel_launch(void* const* d_in, const int* in_sizes, int n_in,
                              void* d_out, int out_size, void* d_ws, size_t ws_size,
                              hipStream_t stream) {
    const float* prev  = (const float*)d_in[0];
    const float* offs  = (const float*)d_in[1];
    const float* dcnw  = (const float*)d_in[2];
    const float* dcnb  = (const float*)d_in[3];
    const float* w1    = (const float*)d_in[4];
    const float* b1    = (const float*)d_in[5];
    const float* w2    = (const float*)d_in[6];
    const float* b2    = (const float*)d_in[7];
    float* out = (float*)d_out;

    char* ws = (char*)d_ws;
    const size_t OFF_ALIGNED = 0;                          // B*C*H*W f32    = 8388608
    const size_t OFF_XP      = 8388608;                    // NHWC f32       = 8388608
    const size_t OFF_X1      = 16777216;                   // padded bf16 128ch = 8652800
    const size_t OFF_X2      = 25430016;                   // padded bf16 64ch  = 4326400
    const size_t OFF_W1F     = 29756416;                   // 147456 B
    const size_t OFF_W2F     = 29903872;                   // 73728 B

    float*          aligned = (float*)(ws + OFF_ALIGNED);
    float*          Xp      = (float*)(ws + OFF_XP);
    unsigned short* X1      = (unsigned short*)(ws + OFF_X1);
    unsigned short* X2      = (unsigned short*)(ws + OFF_X2);
    unsigned short* W1F     = (unsigned short*)(ws + OFF_W1F);
    unsigned short* W2F     = (unsigned short*)(ws + OFF_W2F);

    // 1) zero padded staging tensors (pads must be 0 every call)
    zero32_kernel<<<512, 256, 0, stream>>>((unsigned int*)X1, (long)(BB * HP * WP * 128 / 2));
    zero32_kernel<<<512, 256, 0, stream>>>((unsigned int*)X2, (long)(BB * HP * WP * 64 / 2));

    // 2) prev -> NHWC f32 (DCN sampling) + X1 channels 64..127 (bf16)
    pack_prev_kernel<<<(BB * HH * WW) / 256, 256, 0, stream>>>(prev, Xp, X1);

    // 3) weights -> pre-swizzled WMMA B-fragments (bf16)
    pack_w1_kernel<<<18, 256, 0, stream>>>(w1, W1F);
    pack_w2_kernel<<<9, 256, 0, stream>>>(w2, W2F);

    // 4) deformable conv -> aligned (f32) + X1 channels 0..63 (bf16)
    dcn_kernel<<<(BB * 8 * HH * WW) / 256, 256, 0, stream>>>(Xp, offs, dcnw, dcnb, aligned, X1);

    // 5) conv1 (128->64) WMMA + lrelu -> X2
    conv1_wmma_kernel<<<128, 256, 0, stream>>>(X1, W1F, b1, X2);

    // 6) conv2 (64->64) WMMA + lrelu + residual -> out
    conv2_wmma_kernel<<<128, 256, 0, stream>>>(X2, W2F, b2, aligned, out);

    (void)in_sizes; (void)n_in; (void)out_size; (void)ws_size;
}
